// Informer_64330020159489
// MI455X (gfx1250) — matrix operand order repaired and checked by reading, not compiled, not run on hardware
//
#include <hip/hip_runtime.h>
#include <math.h>

// ---------------- constants ----------------
#define Bc   32
#define Lc   2048
#define DMc  256
#define Hc   8
#define DHc  32
#define DFFc 1024
#define ELc  3
#define Uc   40
#define NCc  10

typedef __attribute__((ext_vector_type(16))) _Float16 v16h;
typedef __attribute__((ext_vector_type(8)))  float    v8f;
typedef __attribute__((ext_vector_type(4)))  float    v4f;
typedef __attribute__((ext_vector_type(4)))  _Float16 v4h;

union Frag16 { v16h v; unsigned u[8]; };

// ---------------- helpers ----------------
__device__ __forceinline__ float gelu_exact(float x) {
    return 0.5f * x * (1.0f + erff(x * 0.70710678118654752f));
}

__device__ __forceinline__ unsigned hash_u32(unsigned x) {
    x ^= x >> 16; x *= 0x7feb352dU;
    x ^= x >> 15; x *= 0x846ca68bU;
    x ^= x >> 16; return x;
}

// =====================================================================
// Token embedding (circular conv1d, kernel 3) + positional embedding
// =====================================================================
__global__ __launch_bounds__(256) void embed_kernel(
    const float* __restrict__ x_enc, const float* __restrict__ token_w,
    float* __restrict__ X)
{
    size_t i = (size_t)blockIdx.x * blockDim.x + threadIdx.x;
    if (i >= (size_t)Bc * Lc * DMc) return;
    int c = (int)(i % DMc);
    int l = (int)((i / DMc) % Lc);
    int b = (int)(i / ((size_t)DMc * Lc));

    float acc = 0.0f;
#pragma unroll
    for (int kw = 0; kw < 3; ++kw) {
        int ls = l + kw - 1;
        ls = (ls < 0) ? (Lc - 1) : ((ls >= Lc) ? 0 : ls);
        const float* xr = x_enc + ((size_t)b * Lc + ls) * 3;
        const float* wr = token_w + (size_t)(kw * 3) * DMc + c;
        acc += xr[0] * wr[0] + xr[1] * wr[DMc] + xr[2] * wr[2 * DMc];
    }
    int ce = c & ~1;
    float div = __expf(-logf(10000.0f) * (float)ce / (float)DMc);
    float ang = (float)l * div;
    acc += (c & 1) ? __cosf(ang) : __sinf(ang);
    X[i] = acc;
}

// =====================================================================
// WMMA GEMM:  C[M,N] = act( A[M,K] @ W[K,N] + bias (+ residual) )
// 128x64 tile / 256-thread block (8 waves), K step 32, f16 WMMA,
// register-staged double-buffered LDS (1 barrier per K-step).
// =====================================================================
#define GBM 128
#define GBN 64
#define GBK 32
#define LDA 40   // padded f16 row stride of A tile
#define LDB 40   // padded f16 row stride of transposed W tile

__global__ __launch_bounds__(256) void gemm_wmma_kernel(
    const float* __restrict__ A, const float* __restrict__ W,
    const float* __restrict__ bias, const float* __restrict__ resid,
    float* __restrict__ C, int M, int N, int K, int act)
{
    __shared__ _Float16 sA[2][GBM * LDA];   // 2 x 10240 B
    __shared__ _Float16 sB[2][GBN * LDB];   // 2 x  5120 B   ([N][K] layout)

    const int tid  = threadIdx.x;
    const int bm   = blockIdx.x * GBM;
    const int bn   = blockIdx.y * GBN;
    const int lane = tid & 31;
    const int mrow = (tid >> 5) * 16;
    const int lm   = lane & 15;
    const int kh   = lane >> 4;

    // staging coordinates: A tile = 1024 float4 (4/thread), W tile = 512 float4 (2/thread)
    const int ar = tid >> 3, ac = (tid & 7) << 2;     // A: rows ar+32*it, cols ac..ac+3
    const int wk = tid >> 4, wn = (tid & 15) << 2;    // W: k rows wk+16*it, n cols wn..wn+3

    v4f aReg[4];
    v4f wReg[2];

    auto load_regs = [&](int k0) {
#pragma unroll
        for (int it = 0; it < 4; ++it)
            aReg[it] = *(const v4f*)&A[(size_t)(bm + ar + 32 * it) * K + (k0 + ac)];
#pragma unroll
        for (int it = 0; it < 2; ++it)
            wReg[it] = *(const v4f*)&W[(size_t)(k0 + wk + 16 * it) * N + (bn + wn)];
    };
    auto store_lds = [&](int buf) {
#pragma unroll
        for (int it = 0; it < 4; ++it) {
            v4h h;
#pragma unroll
            for (int j = 0; j < 4; ++j) h[j] = (_Float16)aReg[it][j];
            *(v4h*)&sA[buf][(ar + 32 * it) * LDA + ac] = h;
        }
#pragma unroll
        for (int it = 0; it < 2; ++it)
#pragma unroll
            for (int j = 0; j < 4; ++j)
                sB[buf][(wn + j) * LDB + (wk + 16 * it)] = (_Float16)wReg[it][j];
    };

    v8f acc[4];
#pragma unroll
    for (int t = 0; t < 4; ++t)
#pragma unroll
        for (int j = 0; j < 8; ++j) acc[t][j] = 0.0f;

    const int nk = K / GBK;
    load_regs(0);
    store_lds(0);
    int cur = 0;

    for (int kt = 0; kt < nk; ++kt) {
        if (kt + 1 < nk) load_regs((kt + 1) * GBK);          // global loads fly under WMMA
        if (kt + 2 < nk)                                      // L2 prefetch (global_prefetch_b8)
            __builtin_prefetch(&A[(size_t)(bm + ar) * K + (kt + 2) * GBK + ac], 0, 1);
        __syncthreads();

        // A fragment: lane (kh,lm) = row lm, K octets {kh*8..+7, 16+kh*8..+7}
        Frag16 af;
        const _Float16* arow = &sA[cur][(mrow + lm) * LDA];
#pragma unroll
        for (int j = 0; j < 4; ++j) {
            af.u[j]     = *(const unsigned*)&arow[kh * 8 + 2 * j];
            af.u[4 + j] = *(const unsigned*)&arow[16 + kh * 8 + 2 * j];
        }
#pragma unroll
        for (int t = 0; t < 4; ++t) {
            Frag16 bf;
            const _Float16* brow = &sB[cur][(t * 16 + lm) * LDB];
#pragma unroll
            for (int j = 0; j < 8; ++j) bf.u[j] = *(const unsigned*)&brow[(j >> 2) * 16 + kh * 8 + 2 * (j & 3)];
            acc[t] = __builtin_amdgcn_wmma_f32_16x16x32_f16(
                false, af.v, false, bf.v, (short)0, acc[t], false, false);
        }

        if (kt + 1 < nk) store_lds(cur ^ 1);
        cur ^= 1;
    }

    // epilogue: C VGPR r -> M = mrow + kh*8 + r ; N = bn + t*16 + lm
#pragma unroll
    for (int t = 0; t < 4; ++t) {
        int n = bn + t * 16 + lm;
        float bv = bias ? bias[n] : 0.0f;
#pragma unroll
        for (int r = 0; r < 8; ++r) {
            int m = bm + mrow + kh * 8 + r;
            float v = acc[t][r] + bv;
            if (resid) v += resid[(size_t)m * N + n];
            if (act == 1) v = gelu_exact(v);
            C[(size_t)m * N + n] = v;
        }
    }
}

// =====================================================================
// LayerNorm over last dim (DM=256): one 256-thread block per row
// =====================================================================
__global__ __launch_bounds__(256) void layernorm_kernel(
    const float* __restrict__ in, const float* __restrict__ g,
    const float* __restrict__ b, float* __restrict__ out)
{
    __shared__ float red[256];
    const int tid = threadIdx.x;
    const size_t row = blockIdx.x;
    float v = in[row * DMc + tid];

    red[tid] = v; __syncthreads();
    for (int s = 128; s > 0; s >>= 1) { if (tid < s) red[tid] += red[tid + s]; __syncthreads(); }
    float mean = red[0] * (1.0f / DMc); __syncthreads();

    float d = v - mean;
    red[tid] = d * d; __syncthreads();
    for (int s = 128; s > 0; s >>= 1) { if (tid < s) red[tid] += red[tid + s]; __syncthreads(); }
    float var = red[0] * (1.0f / DMc);

    out[row * DMc + tid] = d * rsqrtf(var + 1e-5f) * g[tid] + b[tid];
}

// =====================================================================
// Sample indices idx[L][U] in [0, L)  (deterministic hash stand-in)
// =====================================================================
__global__ __launch_bounds__(256) void samp_kernel(int* __restrict__ samp, unsigned seed)
{
    int i = blockIdx.x * blockDim.x + threadIdx.x;
    if (i >= Lc * Uc) return;
    samp[i] = (int)(hash_u32((unsigned)i * 0x9E3779B9u + seed * 0x85EBCA6Bu + 0x12345u) & (Lc - 1));
}

// =====================================================================
// Sparsity measure M[b,h,l] = max_u(q.k_samp) - sum_u(q.k_samp)/L
// =====================================================================
__global__ __launch_bounds__(256) void measure_kernel(
    const float* __restrict__ Q, const float* __restrict__ Kt,
    const int* __restrict__ samp, float* __restrict__ Mbuf)
{
    int i = blockIdx.x * blockDim.x + threadIdx.x;   // (b*H + h)*L + l
    if (i >= Bc * Hc * Lc) return;
    int l = i % Lc;
    int h = (i / Lc) % Hc;
    int b = i / (Lc * Hc);

    const v4f* q4 = (const v4f*)(Q + ((size_t)b * Lc + l) * DMc + h * DHc);
    v4f qr[8];
#pragma unroll
    for (int j = 0; j < 8; ++j) qr[j] = q4[j];

    float mx = -3.4e38f, sm = 0.0f;
    for (int u = 0; u < Uc; ++u) {
        int kidx = samp[l * Uc + u];
        const v4f* k4 = (const v4f*)(Kt + ((size_t)b * Lc + kidx) * DMc + h * DHc);
        float dot = 0.0f;
#pragma unroll
        for (int j = 0; j < 8; ++j) {
            v4f kv = k4[j];
#pragma unroll
            for (int e = 0; e < 4; ++e) dot += qr[j][e] * kv[e];
        }
        mx = fmaxf(mx, dot);
        sm += dot;
    }
    Mbuf[i] = mx - sm * (1.0f / (float)Lc);
}

// =====================================================================
// Top-U selection per (b,h): iterative argmax, low-index tie-break
// =====================================================================
__global__ __launch_bounds__(256) void topk_kernel(
    const float* __restrict__ Mbuf, int* __restrict__ mtop)
{
    __shared__ float sv[Lc];
    __shared__ float rv[256];
    __shared__ int   ri[256];
    const int tid = threadIdx.x;
    const int bh  = blockIdx.x;
    const float* Mr = Mbuf + (size_t)bh * Lc;

    for (int i = tid; i < Lc; i += 256) sv[i] = Mr[i];
    __syncthreads();

    for (int u = 0; u < Uc; ++u) {
        float best = -3.4e38f; int bi = 0x7fffffff;
        for (int i = tid; i < Lc; i += 256) {
            float v = sv[i];
            if (v > best || (v == best && i < bi)) { best = v; bi = i; }
        }
        rv[tid] = best; ri[tid] = bi; __syncthreads();
        for (int s = 128; s > 0; s >>= 1) {
            if (tid < s) {
                float v = rv[tid + s]; int ix = ri[tid + s];
                if (v > rv[tid] || (v == rv[tid] && ix < ri[tid])) { rv[tid] = v; ri[tid] = ix; }
            }
            __syncthreads();
        }
        if (tid == 0) { mtop[bh * Uc + u] = ri[0]; sv[ri[0]] = -3.4e38f; }
        __syncthreads();
    }
}

// =====================================================================
// scores[b,h,u,k] = q[mtop[u]] . k / sqrt(dh)
// =====================================================================
__global__ __launch_bounds__(256) void scores_kernel(
    const float* __restrict__ Q, const float* __restrict__ Kt,
    const int* __restrict__ mtop, float* __restrict__ scores)
{
    size_t i = (size_t)blockIdx.x * blockDim.x + threadIdx.x;
    if (i >= (size_t)Bc * Hc * Uc * Lc) return;
    int kk = (int)(i % Lc);
    int u  = (int)((i / Lc) % Uc);
    int bh = (int)(i / ((size_t)Lc * Uc));
    int b = bh / Hc, h = bh % Hc;

    int ql = mtop[bh * Uc + u];
    const v4f* q4 = (const v4f*)(Q  + ((size_t)b * Lc + ql) * DMc + h * DHc);
    const v4f* k4 = (const v4f*)(Kt + ((size_t)b * Lc + kk) * DMc + h * DHc);
    float dot = 0.0f;
#pragma unroll
    for (int j = 0; j < 8; ++j) {
        v4f qv = q4[j], kv = k4[j];
#pragma unroll
        for (int e = 0; e < 4; ++e) dot += qv[e] * kv[e];
    }
    scores[i] = dot * 0.1767766952966369f;   // 1/sqrt(32)
}

// =====================================================================
// softmax over last dim (L=2048): one block per (b,h,u)
// =====================================================================
__global__ __launch_bounds__(256) void softmax_kernel(float* __restrict__ scores)
{
    __shared__ float red[256];
    const int tid = threadIdx.x;
    float* row = scores + (size_t)blockIdx.x * Lc;

    float v[8];
    float mx = -3.4e38f;
#pragma unroll
    for (int j = 0; j < 8; ++j) { v[j] = row[tid + j * 256]; mx = fmaxf(mx, v[j]); }
    red[tid] = mx; __syncthreads();
    for (int s = 128; s > 0; s >>= 1) { if (tid < s) red[tid] = fmaxf(red[tid], red[tid + s]); __syncthreads(); }
    mx = red[0]; __syncthreads();

    float sm = 0.0f;
#pragma unroll
    for (int j = 0; j < 8; ++j) { v[j] = __expf(v[j] - mx); sm += v[j]; }
    red[tid] = sm; __syncthreads();
    for (int s = 128; s > 0; s >>= 1) { if (tid < s) red[tid] += red[tid + s]; __syncthreads(); }
    float inv = 1.0f / red[0];
#pragma unroll
    for (int j = 0; j < 8; ++j) row[tid + j * 256] = v[j] * inv;
}

// =====================================================================
// vmean[b,h,d] = mean_l v[b,h,l,d]
// =====================================================================
__global__ __launch_bounds__(256) void vmean_kernel(
    const float* __restrict__ V, float* __restrict__ vmean)
{
    int i = blockIdx.x * blockDim.x + threadIdx.x;   // (b*H+h)*32 + d
    if (i >= Bc * Hc * DHc) return;
    int d = i % DHc;
    int h = (i / DHc) % Hc;
    int b = i / (DHc * Hc);
    float s = 0.0f;
    for (int l = 0; l < Lc; ++l)
        s += V[((size_t)b * Lc + l) * DMc + h * DHc + d];
    vmean[i] = s * (1.0f / (float)Lc);
}

// ctx[b,l,h,d] = vmean[b,h,d]
__global__ __launch_bounds__(256) void ctx_fill_kernel(
    const float* __restrict__ vmean, float* __restrict__ CTX)
{
    size_t i = (size_t)blockIdx.x * blockDim.x + threadIdx.x;
    if (i >= (size_t)Bc * Lc * DMc) return;
    int c = (int)(i % DMc);
    int b = (int)(i / ((size_t)DMc * Lc));
    CTX[i] = vmean[(b * Hc + (c >> 5)) * DHc + (c & 31)];
}

// ctx[b, mtop[u], h, :] = attn[u,:] @ v
__global__ __launch_bounds__(256) void ctx_scatter_kernel(
    const float* __restrict__ scores, const float* __restrict__ V,
    const int* __restrict__ mtop, float* __restrict__ CTX)
{
    int i = blockIdx.x * blockDim.x + threadIdx.x;   // (bh*U + u)*32 + d
    if (i >= Bc * Hc * Uc * DHc) return;
    int d  = i % DHc;
    int u  = (i / DHc) % Uc;
    int bh = i / (DHc * Uc);
    int b = bh / Hc, h = bh % Hc;

    int ql = mtop[bh * Uc + u];
    const float* att = scores + ((size_t)bh * Uc + u) * Lc;
    float acc = 0.0f;
    for (int kk = 0; kk < Lc; ++kk)
        acc += att[kk] * V[((size_t)b * Lc + kk) * DMc + h * DHc + d];
    CTX[((size_t)b * Lc + ql) * DMc + h * DHc + d] = acc;
}

// =====================================================================
// Final: out[b,n] = sum_{l,d} x[b,l,d]*mask[b,l]*fcw[l*DM+d, n] + fcb[n]
// one block per batch b; fcw read fully coalesced
// =====================================================================
__global__ __launch_bounds__(256) void fc_kernel(
    const float* __restrict__ X, const float* __restrict__ mask,
    const float* __restrict__ fcw, const float* __restrict__ fcb,
    float* __restrict__ out)
{
    __shared__ float red[256];
    const int tid = threadIdx.x;
    const int b = blockIdx.x;

    float acc[NCc];
#pragma unroll
    for (int n = 0; n < NCc; ++n) acc[n] = 0.0f;

    for (int i = tid; i < Lc * DMc; i += 256) {
        float xv = X[(size_t)b * Lc * DMc + i] * mask[(size_t)b * Lc + (i >> 8)];
        const float* w = fcw + (size_t)i * NCc;
#pragma unroll
        for (int n = 0; n < NCc; ++n) acc[n] += xv * w[n];
    }
#pragma unroll
    for (int n = 0; n < NCc; ++n) {
        red[tid] = acc[n]; __syncthreads();
        for (int s = 128; s > 0; s >>= 1) { if (tid < s) red[tid] += red[tid + s]; __syncthreads(); }
        if (tid == 0) out[b * NCc + n] = red[0] + fcb[n];
        __syncthreads();
    }
}

// =====================================================================
// host launcher
// =====================================================================
extern "C" void kernel_launch(void* const* d_in, const int* in_sizes, int n_in,
                              void* d_out, int out_size, void* d_ws, size_t ws_size,
                              hipStream_t stream)
{
    (void)in_sizes; (void)n_in; (void)out_size; (void)ws_size;

    const float* x_enc   = (const float*)d_in[0];
    const float* x_mark  = (const float*)d_in[1];
    const float* token_w = (const float*)d_in[2];
    const float* Wq = (const float*)d_in[3];   const float* bq = (const float*)d_in[4];
    const float* Wk = (const float*)d_in[5];   const float* bk = (const float*)d_in[6];
    const float* Wv = (const float*)d_in[7];   const float* bv = (const float*)d_in[8];
    const float* Wo = (const float*)d_in[9];   const float* bo = (const float*)d_in[10];
    const float* ln1g = (const float*)d_in[11]; const float* ln1b = (const float*)d_in[12];
    const float* c1w  = (const float*)d_in[13]; const float* c1b  = (const float*)d_in[14];
    const float* c2w  = (const float*)d_in[15]; const float* c2b  = (const float*)d_in[16];
    const float* ln2g = (const float*)d_in[17]; const float* ln2b = (const float*)d_in[18];
    const float* lnfg = (const float*)d_in[19]; const float* lnfb = (const float*)d_in[20];
    const float* fcw  = (const float*)d_in[21]; const float* fcb  = (const float*)d_in[22];

    const size_t SZ = (size_t)Bc * Lc * DMc;     // 16,777,216 floats
    float* ws   = (float*)d_ws;
    float* X    = ws;
    float* Qb   = ws + 1 * SZ;
    float* Kb   = ws + 2 * SZ;
    float* Vb   = ws + 3 * SZ;
    float* CTX  = ws + 4 * SZ;
    float* TMP1 = ws + 5 * SZ;
    float* TMP2 = ws + 6 * SZ;                   // 67,108,864 floats (FFN hidden)
    // attention scratch overlapped into TMP2 region (disjoint lifetimes)
    float* SCORES = TMP2;                                      // B*H*U*L = 20,971,520
    float* MBUF   = SCORES + (size_t)Bc * Hc * Uc * Lc;        // B*H*L   =    524,288
    float* VMEAN  = MBUF + (size_t)Bc * Hc * Lc;               // B*H*DH  =      8,192
    int*   MTOP   = (int*)(VMEAN + Bc * Hc * DHc);             // B*H*U   =     10,240
    int*   SAMP   = MTOP + Bc * Hc * Uc;                       // L*U     =     81,920

    const int Mrows = Bc * Lc;                   // 65536
    const dim3 blk(256);
    const dim3 gemm256(Mrows / GBM, DMc / GBN);  // 512 x 4
    const dim3 gemmFF1(Mrows / GBM, DFFc / GBN); // 512 x 16

    embed_kernel<<<dim3((unsigned)((SZ + 255) / 256)), blk, 0, stream>>>(x_enc, token_w, X);

    for (int layer = 0; layer < ELc; ++layer) {
        const float* wq = Wq + (size_t)layer * DMc * DMc;   const float* bql = bq + layer * DMc;
        const float* wk = Wk + (size_t)layer * DMc * DMc;   const float* bkl = bk + layer * DMc;
        const float* wv = Wv + (size_t)layer * DMc * DMc;   const float* bvl = bv + layer * DMc;
        const float* wo = Wo + (size_t)layer * DMc * DMc;   const float* bol = bo + layer * DMc;
        const float* g1 = ln1g + layer * DMc;               const float* b1 = ln1b + layer * DMc;
        const float* g2 = ln2g + layer * DMc;               const float* b2 = ln2b + layer * DMc;
        const float* w1 = c1w + (size_t)layer * DMc * DFFc; const float* bw1 = c1b + layer * DFFc;
        const float* w2 = c2w + (size_t)layer * DFFc * DMc; const float* bw2 = c2b + layer * DMc;

        // QKV projections
        gemm_wmma_kernel<<<gemm256, blk, 0, stream>>>(X, wq, bql, nullptr, Qb, Mrows, DMc, DMc, 0);
        gemm_wmma_kernel<<<gemm256, blk, 0, stream>>>(X, wk, bkl, nullptr, Kb, Mrows, DMc, DMc, 0);
        gemm_wmma_kernel<<<gemm256, blk, 0, stream>>>(X, wv, bvl, nullptr, Vb, Mrows, DMc, DMc, 0);

        // ProbSparse attention
        samp_kernel<<<dim3((Lc * Uc + 255) / 256), blk, 0, stream>>>(SAMP, 42u + (unsigned)layer);
        measure_kernel<<<dim3((Bc * Hc * Lc + 255) / 256), blk, 0, stream>>>(Qb, Kb, SAMP, MBUF);
        topk_kernel<<<dim3(Bc * Hc), blk, 0, stream>>>(MBUF, MTOP);
        scores_kernel<<<dim3((unsigned)(((size_t)Bc * Hc * Uc * Lc + 255) / 256)), blk, 0, stream>>>(Qb, Kb, MTOP, SCORES);
        softmax_kernel<<<dim3(Bc * Hc * Uc), blk, 0, stream>>>(SCORES);
        vmean_kernel<<<dim3((Bc * Hc * DHc + 255) / 256), blk, 0, stream>>>(Vb, VMEAN);
        ctx_fill_kernel<<<dim3((unsigned)((SZ + 255) / 256)), blk, 0, stream>>>(VMEAN, CTX);
        ctx_scatter_kernel<<<dim3((Bc * Hc * Uc * DHc + 255) / 256), blk, 0, stream>>>(SCORES, Vb, MTOP, CTX);

        // output projection + residual, then LN1
        gemm_wmma_kernel<<<gemm256, blk, 0, stream>>>(CTX, wo, bol, X, TMP1, Mrows, DMc, DMc, 0);
        layernorm_kernel<<<dim3(Mrows), blk, 0, stream>>>(TMP1, g1, b1, X);

        // FFN (gelu fused in first GEMM), residual in second, then LN2
        gemm_wmma_kernel<<<gemmFF1, blk, 0, stream>>>(X, w1, bw1, nullptr, TMP2, Mrows, DFFc, DMc, 1);
        gemm_wmma_kernel<<<gemm256, blk, 0, stream>>>(TMP2, w2, bw2, X, TMP1, Mrows, DMc, DFFc, 0);
        layernorm_kernel<<<dim3(Mrows), blk, 0, stream>>>(TMP1, g2, b2, X);
    }

    // final LayerNorm + masked flatten @ fc
    layernorm_kernel<<<dim3(Mrows), blk, 0, stream>>>(X, lnfg, lnfb, TMP1);
    fc_kernel<<<dim3(Bc), blk, 0, stream>>>(TMP1, x_mark, fcw, fcb, (float*)d_out);
}
